// Model_32195074850987
// MI455X (gfx1250) — compile-verified, matrix operand order checked by reading, not run on hardware
//
#include <hip/hip_runtime.h>

#define MTOT 8192
#define EPSV 1e-5f
#define ATT_SCALE 0.17677669529663689f

typedef __attribute__((ext_vector_type(16))) _Float16 v16h;
typedef __attribute__((ext_vector_type(8)))  _Float16 v8h;
typedef __attribute__((ext_vector_type(8)))  float    v8f;

__device__ __forceinline__ float softplus_f(float x) {
  return x > 20.f ? x : log1pf(__expf(x));
}

// Convert two float4s (8 consecutive f32) into 8 f16 slots of a fragment.
__device__ __forceinline__ void cvt8(v16h& dst, int base, float4 a, float4 b) {
  dst[base + 0] = (_Float16)a.x; dst[base + 1] = (_Float16)a.y;
  dst[base + 2] = (_Float16)a.z; dst[base + 3] = (_Float16)a.w;
  dst[base + 4] = (_Float16)b.x; dst[base + 5] = (_Float16)b.y;
  dst[base + 6] = (_Float16)b.z; dst[base + 7] = (_Float16)b.w;
}

__device__ __forceinline__ void zero8(v16h& dst, int base) {
#pragma unroll
  for (int j = 0; j < 8; ++j) dst[base + j] = (_Float16)0.f;
}

// One-shot f32 -> f16 weight conversion (weights are L2-resident and reused
// by all 8192 token rows; this halves weight-load bytes in every GEMM).
__global__ __launch_bounds__(256) void cvt_w_kernel(
    const float* __restrict__ src, _Float16* __restrict__ dst, int n)
{
  int i = blockIdx.x * 256 + threadIdx.x;
  if (i < n) dst[i] = (_Float16)src[i];
}

// ---------------------------------------------------------------------------
// WMMA GEMM:  C[M,N] = act(A[M,K] @ W[N,K]^T + bias)   (optional +=)
// W is pre-converted f16. Each wave owns a 16(M) x 64(N) strip: one A
// fragment feeds 4 back-to-back v_wmma_f32_16x16x32_f16 per 32-wide K step.
// Main K loop is branch-free (A: 4x b128 + cvt, W: 2x b128 straight f16 per
// tile); a single 16-wide tail handles K%32==16 (the K=16 dt projection).
// W row pointers are clamped; the masked epilogue discards junk columns.
// ---------------------------------------------------------------------------
__global__ __launch_bounds__(256) void wmma_gemm_kernel(
    const float* __restrict__ A, int lda,
    const _Float16* __restrict__ W,      // N x K row-major, f16
    const float* __restrict__ bias,      // N (nullable)
    float* __restrict__ C, int ldc,
    int M, int N, int K, int act, int resid)
{
  const int wave = threadIdx.x >> 5;
  const int lane = threadIdx.x & 31;
  const int ntiles = (N + 15) >> 4;
  const int ntblk  = (ntiles + 3) >> 2;
  const int mtiles = M >> 4;
  const int tile = blockIdx.x * 8 + wave;
  if (tile >= mtiles * ntblk) return;
  const int mt  = tile / ntblk;
  const int ntb = tile - mt * ntblk;
  const int r16 = lane & 15;
  const int half = lane >> 4;

  const float* __restrict__ Arow = A + (size_t)(mt * 16 + r16) * lda + half * 8;
  const _Float16* __restrict__ Wrow[4];
  int wcol[4];
#pragma unroll
  for (int t = 0; t < 4; ++t) {
    int c = (ntb * 4 + t) * 16 + r16;
    wcol[t] = c;
    int cc = c < N ? c : (N - 1);           // clamp: junk cols masked at store
    Wrow[t] = W + (size_t)cc * K + half * 8;
  }

  v8f acc[4] = {{}, {}, {}, {}};
  const int kfull = K & ~31;
  int k0 = 0;
  for (; k0 < kfull; k0 += 32) {
    float4 a0 = *(const float4*)(Arow + k0);
    float4 a1 = *(const float4*)(Arow + k0 + 4);
    float4 a2 = *(const float4*)(Arow + k0 + 16);
    float4 a3 = *(const float4*)(Arow + k0 + 20);
    v8h w0[4], w1[4];
#pragma unroll
    for (int t = 0; t < 4; ++t) {
      w0[t] = *(const v8h*)(Wrow[t] + k0);
      w1[t] = *(const v8h*)(Wrow[t] + k0 + 16);
    }
    v16h af;
    cvt8(af, 0, a0, a1);
    cvt8(af, 8, a2, a3);
#pragma unroll
    for (int t = 0; t < 4; ++t) {
      v16h bf = __builtin_shufflevector(w0[t], w1[t],
                                        0, 1, 2, 3, 4, 5, 6, 7,
                                        8, 9, 10, 11, 12, 13, 14, 15);
      acc[t] = __builtin_amdgcn_wmma_f32_16x16x32_f16(false, af, false, bf,
                                                      (short)0, acc[t], false, false);
    }
  }
  if (k0 < K) {                             // K % 32 == 16 tail (dt proj only)
    v16h af;
    cvt8(af, 0, *(const float4*)(Arow + k0), *(const float4*)(Arow + k0 + 4));
    zero8(af, 8);
    const v8h zz = {};
#pragma unroll
    for (int t = 0; t < 4; ++t) {
      v8h w0 = *(const v8h*)(Wrow[t] + k0);
      v16h bf = __builtin_shufflevector(w0, zz,
                                        0, 1, 2, 3, 4, 5, 6, 7,
                                        8, 9, 10, 11, 12, 13, 14, 15);
      acc[t] = __builtin_amdgcn_wmma_f32_16x16x32_f16(false, af, false, bf,
                                                      (short)0, acc[t], false, false);
    }
  }

#pragma unroll
  for (int t = 0; t < 4; ++t) {
    if (wcol[t] < N) {
      const float bval = bias ? bias[wcol[t]] : 0.f;
#pragma unroll
      for (int r = 0; r < 8; ++r) {
        int row = mt * 16 + r + 8 * half;   // C layout: M=r+8*half, N=lane&15
        float v = acc[t][r] + bval;
        if (act == 1) v = softplus_f(v);
        size_t o = (size_t)row * ldc + wcol[t];
        if (resid) C[o] += v; else C[o] = v;
      }
    }
  }
}

// ---------------------------------------------------------------------------
// LayerNorm / RMSNorm: one wave per 256-wide row, xor-shuffle reductions.
// ---------------------------------------------------------------------------
__global__ __launch_bounds__(256) void ln_kernel(
    const float* __restrict__ X, const float* __restrict__ w,
    const float* __restrict__ b, float* __restrict__ Y, int M)
{
  int row = blockIdx.x * 8 + (threadIdx.x >> 5);
  if (row >= M) return;
  int lane = threadIdx.x & 31;
  const float* x = X + (size_t)row * 256;
  float v[8]; float s = 0.f;
#pragma unroll
  for (int i = 0; i < 8; ++i) { v[i] = x[lane + 32 * i]; s += v[i]; }
#pragma unroll
  for (int m = 16; m >= 1; m >>= 1) s += __shfl_xor(s, m, 32);
  float mu = s * (1.f / 256.f);
  float s2 = 0.f;
#pragma unroll
  for (int i = 0; i < 8; ++i) { float d = v[i] - mu; s2 += d * d; }
#pragma unroll
  for (int m = 16; m >= 1; m >>= 1) s2 += __shfl_xor(s2, m, 32);
  float rs = rsqrtf(s2 * (1.f / 256.f) + EPSV);
  float* y = Y + (size_t)row * 256;
#pragma unroll
  for (int i = 0; i < 8; ++i) {
    int c = lane + 32 * i;
    y[c] = (v[i] - mu) * rs * w[c] + b[c];
  }
}

__global__ __launch_bounds__(256) void rms_kernel(
    const float* __restrict__ X, const float* __restrict__ w,
    float* __restrict__ Y, int M)
{
  int row = blockIdx.x * 8 + (threadIdx.x >> 5);
  if (row >= M) return;
  int lane = threadIdx.x & 31;
  const float* x = X + (size_t)row * 256;
  float v[8]; float s2 = 0.f;
#pragma unroll
  for (int i = 0; i < 8; ++i) { v[i] = x[lane + 32 * i]; s2 += v[i] * v[i]; }
#pragma unroll
  for (int m = 16; m >= 1; m >>= 1) s2 += __shfl_xor(s2, m, 32);
  float rs = rsqrtf(s2 * (1.f / 256.f) + EPSV);
  float* y = Y + (size_t)row * 256;
#pragma unroll
  for (int i = 0; i < 8; ++i) {
    int c = lane + 32 * i;
    y[c] = v[i] * rs * w[c];
  }
}

// ---------------------------------------------------------------------------
// Depthwise causal conv (width 4) over time + bias + SiLU.
// ---------------------------------------------------------------------------
__global__ __launch_bounds__(256) void conv_silu_kernel(
    const float* __restrict__ xz, const float* __restrict__ cw,
    const float* __restrict__ cb, float* __restrict__ out)
{
  int idx = blockIdx.x * 256 + threadIdx.x;
  if (idx >= MTOT * 512) return;
  int d = idx & 511;
  int row = idx >> 9;
  int t = row & 2047;
  float acc = cb[d];
#pragma unroll
  for (int j = 0; j < 4; ++j) {
    int tt = t + j - 3;
    if (tt >= 0) acc += cw[d * 4 + j] * xz[(size_t)(row + j - 3) * 1024 + d];
  }
  out[(size_t)row * 512 + d] = acc / (1.f + __expf(-acc));
}

// ---------------------------------------------------------------------------
// Selective scan: each 16-lane half-wave owns one (b,d) channel; lane n holds
// state n. y reduced over states with xor-shuffles (masks stay in the half).
// Fuses +xm*D and *silu(z).
// ---------------------------------------------------------------------------
__global__ __launch_bounds__(256) void scan_kernel(
    const float* __restrict__ delta, const float* __restrict__ xmc,
    const float* __restrict__ dbc, const float* __restrict__ xz,
    const float* __restrict__ A_log, const float* __restrict__ Dp,
    float* __restrict__ yv)
{
  int gid = blockIdx.x * 16 + (threadIdx.x >> 4);   // over 4*512 channels
  int n = threadIdx.x & 15;
  if (gid >= 2048) return;
  int b = gid >> 9, d = gid & 511;
  float a = -__expf(A_log[d * 16 + n]);
  float Dd = Dp[d];
  float h = 0.f;
  size_t base = (size_t)b * 2048;
  for (int t = 0; t < 2048; ++t) {
    size_t row = base + t;
    float dv = delta[row * 512 + d];
    float xv = xmc[row * 512 + d];
    float Bv = dbc[row * 48 + 16 + n];
    float Cv = dbc[row * 48 + 32 + n];
    h = __expf(dv * a) * h + dv * Bv * xv;
    float c = h * Cv;
    c += __shfl_xor(c, 1, 32);
    c += __shfl_xor(c, 2, 32);
    c += __shfl_xor(c, 4, 32);
    c += __shfl_xor(c, 8, 32);
    if (n == 0) {
      float z = xz[row * 1024 + 512 + d];
      float sil = z / (1.f + __expf(-z));
      yv[row * 512 + d] = (c + xv * Dd) * sil;
    }
  }
}

// ---------------------------------------------------------------------------
// Flash attention, one wave per (b, head, 16-query tile). head=32 => QK^T is
// one v_wmma per 16-key tile; PV accumulates in 32-key chunks. P goes
// C-layout -> LDS -> A-layout (explicit s_wait_dscnt fence).
// ---------------------------------------------------------------------------
__global__ __launch_bounds__(256) void attn_kernel(
    const float* __restrict__ qkv, float* __restrict__ yo)
{
  __shared__ _Float16 pt[8][16][40];
  const int wave = threadIdx.x >> 5;
  const int lane = threadIdx.x & 31;
  const int gwave = blockIdx.x * 8 + wave;
  if (gwave >= 4096) return;
  const int qt = gwave & 127;
  const int bh = gwave >> 7;
  const int hI = bh & 7;
  const int b  = bh >> 3;
  const int r16 = lane & 15;
  const int half = lane >> 4;
  const size_t bbase = (size_t)b * 2048;
  const size_t hb = (size_t)hI * 96;     // qkv row: h*96 + which*32 + d
  const int qbase = qt * 16;

  v16h aQ;
  {
    const float* qp = qkv + (bbase + qbase + r16) * 768 + hb + half * 8;
    cvt8(aQ, 0, *(const float4*)qp, *(const float4*)(qp + 4));
    cvt8(aQ, 8, *(const float4*)(qp + 16), *(const float4*)(qp + 20));
  }

  float m8[8], l8[8];
  v8f acc0 = {}, acc1 = {};
#pragma unroll
  for (int r = 0; r < 8; ++r) { m8[r] = -3.0e38f; l8[r] = 0.f; }

  const int nkb = (qbase + 15) / 32 + 1;
  for (int kb = 0; kb < nkb; ++kb) {
    const int s0 = kb * 32;
    v8f sc0, sc1;
    {
      v16h bK;
      const float* kp = qkv + (bbase + s0 + r16) * 768 + hb + 32 + half * 8;
      cvt8(bK, 0, *(const float4*)kp, *(const float4*)(kp + 4));
      cvt8(bK, 8, *(const float4*)(kp + 16), *(const float4*)(kp + 20));
      v8f z = {};
      sc0 = __builtin_amdgcn_wmma_f32_16x16x32_f16(false, aQ, false, bK, (short)0, z, false, false);
    }
    {
      v16h bK;
      const float* kp = qkv + (bbase + s0 + 16 + r16) * 768 + hb + 32 + half * 8;
      cvt8(bK, 0, *(const float4*)kp, *(const float4*)(kp + 4));
      cvt8(bK, 8, *(const float4*)(kp + 16), *(const float4*)(kp + 20));
      v8f z = {};
      sc1 = __builtin_amdgcn_wmma_f32_16x16x32_f16(false, aQ, false, bK, (short)0, z, false, false);
    }
    // causal mask + online softmax (per-row stats via half-wave xor reductions)
#pragma unroll
    for (int r = 0; r < 8; ++r) {
      const int q = qbase + r + 8 * half;
      float v0 = (s0 + r16      <= q) ? sc0[r] * ATT_SCALE : -3.0e38f;
      float v1 = (s0 + 16 + r16 <= q) ? sc1[r] * ATT_SCALE : -3.0e38f;
      float mb = fmaxf(v0, v1);
      mb = fmaxf(mb, __shfl_xor(mb, 1, 32));
      mb = fmaxf(mb, __shfl_xor(mb, 2, 32));
      mb = fmaxf(mb, __shfl_xor(mb, 4, 32));
      mb = fmaxf(mb, __shfl_xor(mb, 8, 32));
      float mnew = fmaxf(m8[r], mb);
      float rescale = __expf(m8[r] - mnew);
      float p0 = __expf(v0 - mnew);
      float p1 = __expf(v1 - mnew);
      float ps = p0 + p1;
      ps += __shfl_xor(ps, 1, 32);
      ps += __shfl_xor(ps, 2, 32);
      ps += __shfl_xor(ps, 4, 32);
      ps += __shfl_xor(ps, 8, 32);
      l8[r] = l8[r] * rescale + ps;
      m8[r] = mnew;
      acc0[r] *= rescale;
      acc1[r] *= rescale;
      pt[wave][r + 8 * half][r16]      = (_Float16)p0;
      pt[wave][r + 8 * half][16 + r16] = (_Float16)p1;
    }
    asm volatile("s_wait_dscnt 0" ::: "memory");
    v16h aP;
    {
      const v8h p0v = *(const v8h*)&pt[wave][r16][half * 8];
      const v8h p1v = *(const v8h*)&pt[wave][r16][16 + half * 8];
      aP = __builtin_shufflevector(p0v, p1v,
                                   0, 1, 2, 3, 4, 5, 6, 7,
                                   8, 9, 10, 11, 12, 13, 14, 15);
    }
#pragma unroll
    for (int nt2 = 0; nt2 < 2; ++nt2) {
      v16h bV;
#pragma unroll
      for (int g = 0; g < 2; ++g)
#pragma unroll
        for (int j = 0; j < 8; ++j) {
          int klocal = g * 16 + half * 8 + j;
          bV[g * 8 + j] = (_Float16)qkv[(bbase + s0 + klocal) * 768 + hb + 64 + nt2 * 16 + r16];
        }
      if (nt2 == 0)
        acc0 = __builtin_amdgcn_wmma_f32_16x16x32_f16(false, aP, false, bV, (short)0, acc0, false, false);
      else
        acc1 = __builtin_amdgcn_wmma_f32_16x16x32_f16(false, aP, false, bV, (short)0, acc1, false, false);
    }
  }
#pragma unroll
  for (int r = 0; r < 8; ++r) {
    const int q = qbase + r + 8 * half;
    float inv = 1.f / l8[r];
    size_t o = (bbase + q) * 256 + (size_t)hI * 32;
    yo[o + r16]      = acc0[r] * inv;
    yo[o + 16 + r16] = acc1[r] * inv;
  }
}

__global__ __launch_bounds__(256) void add_kernel(
    float* __restrict__ dst, const float* __restrict__ src, int n)
{
  int i = blockIdx.x * 256 + threadIdx.x;
  if (i < n) dst[i] += src[i];
}

// ---------------------------------------------------------------------------
// Host orchestration
// ---------------------------------------------------------------------------
struct MambaP { const float *rms_w,*in_proj,*conv_w,*conv_b,*x_proj,*dt_w,*dt_b,*A_log,*Dp,*out_proj; };
struct BlockP {
  const float *ln_m_w,*ln_m_b,*ln_a_w,*ln_a_b,*ln_f_w,*ln_f_b;
  const float *attn_w,*attn_b,*proj_w,*proj_b,*mlp_w,*mlp_b;
  MambaP m[2];
};
struct NetP { const float* x; const float *W_in,*b_in,*W_dec,*b_dec; BlockP blk[2]; };

static inline const float* FP(void* p) { return (const float*)p; }

static void fill_insertion(NetP& n, void* const* p) {
  int i = 0;
  n.x = FP(p[i++]);
  n.W_in = FP(p[i++]); n.b_in = FP(p[i++]);
  for (int bi = 0; bi < 2; ++bi) {
    BlockP& B = n.blk[bi];
    B.ln_m_w=FP(p[i++]); B.ln_m_b=FP(p[i++]);
    B.ln_a_w=FP(p[i++]); B.ln_a_b=FP(p[i++]);
    B.ln_f_w=FP(p[i++]); B.ln_f_b=FP(p[i++]);
    B.attn_w=FP(p[i++]); B.attn_b=FP(p[i++]);
    B.proj_w=FP(p[i++]); B.proj_b=FP(p[i++]);
    B.mlp_w =FP(p[i++]); B.mlp_b =FP(p[i++]);
    for (int l = 0; l < 2; ++l) {
      MambaP& m = B.m[l];
      m.rms_w=FP(p[i++]); m.in_proj=FP(p[i++]); m.conv_w=FP(p[i++]); m.conv_b=FP(p[i++]);
      m.x_proj=FP(p[i++]); m.dt_w=FP(p[i++]); m.dt_b=FP(p[i++]); m.A_log=FP(p[i++]);
      m.Dp=FP(p[i++]); m.out_proj=FP(p[i++]);
    }
  }
  n.W_dec=FP(p[i++]); n.b_dec=FP(p[i++]);
}

static int fill_params_sorted(NetP& n, void* const* p, int i) {
  n.W_dec=FP(p[i++]); n.W_in=FP(p[i++]); n.b_dec=FP(p[i++]); n.b_in=FP(p[i++]);
  for (int bi = 0; bi < 2; ++bi) {
    BlockP& B = n.blk[bi];
    B.attn_b=FP(p[i++]); B.attn_w=FP(p[i++]);
    B.ln_a_b=FP(p[i++]); B.ln_a_w=FP(p[i++]);
    B.ln_f_b=FP(p[i++]); B.ln_f_w=FP(p[i++]);
    B.ln_m_b=FP(p[i++]); B.ln_m_w=FP(p[i++]);
    for (int l = 0; l < 2; ++l) {
      MambaP& m = B.m[l];
      m.A_log=FP(p[i++]); m.Dp=FP(p[i++]); m.conv_b=FP(p[i++]); m.conv_w=FP(p[i++]);
      m.dt_b=FP(p[i++]); m.dt_w=FP(p[i++]); m.in_proj=FP(p[i++]); m.out_proj=FP(p[i++]);
      m.rms_w=FP(p[i++]); m.x_proj=FP(p[i++]);
    }
    B.mlp_b=FP(p[i++]); B.mlp_w=FP(p[i++]); B.proj_b=FP(p[i++]); B.proj_w=FP(p[i++]);
  }
  return i;
}

extern "C" void kernel_launch(void* const* d_in, const int* in_sizes, int n_in,
                              void* d_out, int out_size, void* d_ws, size_t ws_size,
                              hipStream_t stream)
{
  (void)out_size; (void)ws_size;
  NetP net;
  if (n_in >= 1 && in_sizes[0] == 2048) {                 // fully sorted pytree
    int i = fill_params_sorted(net, d_in, 0);
    net.x = FP(d_in[i]);
  } else if (n_in >= 2 && in_sizes[1] == 2048) {          // x first, params sorted
    net.x = FP(d_in[0]);
    fill_params_sorted(net, d_in, 1);
  } else {                                                // dict insertion order
    fill_insertion(net, d_in);
  }

  const int M = MTOT;
  float* wsf   = (float*)d_ws;
  float* hbuf  = wsf;                       // M*256
  float* ubuf  = hbuf  + (size_t)M * 256;   // M*256
  float* lnbuf = ubuf  + (size_t)M * 256;   // M*256
  float* xz    = lnbuf + (size_t)M * 256;   // M*1024 (also qkv: M*768)
  float* xmc   = xz    + (size_t)M * 1024;  // M*512  (also y_attn: M*256)
  float* delta = xmc   + (size_t)M * 512;   // M*512
  float* yv    = delta + (size_t)M * 512;   // M*512
  float* dbc   = yv    + (size_t)M * 512;   // M*48

  // f16 weight pool lives after the f32 scratch (16B-aligned by construction)
  _Float16* whp = (_Float16*)(dbc + (size_t)M * 48);
  size_t woff = 0;
  auto cvtw = [&](const float* src, size_t n) -> const _Float16* {
    _Float16* dst = whp + woff; woff += n;
    cvt_w_kernel<<<(int)((n + 255) / 256), 256, 0, stream>>>(src, dst, (int)n);
    return dst;
  };

  // Convert every GEMM weight once (deterministic, replayed per launch).
  const _Float16* W_in_h  = cvtw(net.W_in, 256 * 32);
  const _Float16* W_dec_h = cvtw(net.W_dec, 8 * 256);
  const _Float16 *attn_h[2], *proj_h[2], *mlp_h[2];
  const _Float16 *inp_h[2][2], *xp_h[2][2], *dtw_h[2][2], *outp_h[2][2];
  for (int bi = 0; bi < 2; ++bi) {
    attn_h[bi] = cvtw(net.blk[bi].attn_w, 768 * 256);
    proj_h[bi] = cvtw(net.blk[bi].proj_w, 256 * 256);
    mlp_h[bi]  = cvtw(net.blk[bi].mlp_w,  256 * 256);
    for (int l = 0; l < 2; ++l) {
      inp_h[bi][l]  = cvtw(net.blk[bi].m[l].in_proj,  1024 * 256);
      xp_h[bi][l]   = cvtw(net.blk[bi].m[l].x_proj,   48 * 512);
      dtw_h[bi][l]  = cvtw(net.blk[bi].m[l].dt_w,     512 * 16);
      outp_h[bi][l] = cvtw(net.blk[bi].m[l].out_proj, 256 * 512);
    }
  }

  auto gemm = [&](const float* A, int lda, const _Float16* W, const float* bias,
                  float* C, int ldc, int N, int K, int act, int resid) {
    int ntiles = (N + 15) / 16;
    int ntblk  = (ntiles + 3) / 4;
    int tiles  = (M / 16) * ntblk;
    int blocks = (tiles + 7) / 8;
    wmma_gemm_kernel<<<blocks, 256, 0, stream>>>(A, lda, W, bias, C, ldc, M, N, K, act, resid);
  };
  const int rows8 = M / 8;

  // Embedding: h = x @ W_in.T + b_in
  gemm(net.x, 32, W_in_h, net.b_in, hbuf, 256, 256, 32, 0, 0);

  for (int bi = 0; bi < 2; ++bi) {
    const BlockP& B = net.blk[bi];

    // u = LN(h)
    ln_kernel<<<rows8, 256, 0, stream>>>(hbuf, B.ln_m_w, B.ln_m_b, ubuf, M);

    for (int l = 0; l < 2; ++l) {
      const MambaP& mp = B.m[l];
      rms_kernel<<<rows8, 256, 0, stream>>>(ubuf, mp.rms_w, lnbuf, M);
      gemm(lnbuf, 256, inp_h[bi][l], nullptr, xz, 1024, 1024, 256, 0, 0);
      conv_silu_kernel<<<(M * 512 + 255) / 256, 256, 0, stream>>>(xz, mp.conv_w, mp.conv_b, xmc);
      gemm(xmc, 512, xp_h[bi][l], nullptr, dbc, 48, 48, 512, 0, 0);
      gemm(dbc, 48, dtw_h[bi][l], mp.dt_b, delta, 512, 512, 16, 1, 0);  // softplus
      scan_kernel<<<128, 256, 0, stream>>>(delta, xmc, dbc, xz, mp.A_log, mp.Dp, yv);
      gemm(yv, 512, outp_h[bi][l], nullptr, ubuf, 256, 256, 512, 0, 1); // u += out
    }

    // h = u + h   (out = u + x)
    add_kernel<<<(M * 256 + 255) / 256, 256, 0, stream>>>(hbuf, ubuf, M * 256);

    // Attention
    ln_kernel<<<rows8, 256, 0, stream>>>(hbuf, B.ln_a_w, B.ln_a_b, lnbuf, M);
    gemm(lnbuf, 256, attn_h[bi], B.attn_b, xz, 768, 768, 256, 0, 0);   // qkv
    attn_kernel<<<512, 256, 0, stream>>>(xz, xmc);                      // y_attn in xmc
    gemm(xmc, 256, proj_h[bi], B.proj_b, hbuf, 256, 256, 256, 0, 1);   // h += proj

    // MLP (LN + linear residual)
    ln_kernel<<<rows8, 256, 0, stream>>>(hbuf, B.ln_f_w, B.ln_f_b, lnbuf, M);
    gemm(lnbuf, 256, mlp_h[bi], B.mlp_b, hbuf, 256, 256, 256, 0, 1);   // h += mlp
  }

  // Decode: out = h @ W_dec.T + b_dec   (N=8, masked 16-wide tile)
  gemm(hbuf, 256, W_dec_h, net.b_dec, (float*)d_out, 8, 8, 256, 0, 0);
}
